// SOABlock_32804960207037
// MI455X (gfx1250) — matrix operand order repaired and checked by reading, not compiled, not run on hardware
//
#include <hip/hip_runtime.h>
#include <hip/hip_bf16.h>

// ---------------------------------------------------------------------------
// Self-attention block (SAGAN) for MI455X / gfx1250.
// All GEMM work on v_wmma_f32_16x16x32_bf16 (f32 accumulate). Flash-attention
// inner loop: K/V tiles staged in block-shared LDS by the Tensor Data Mover
// (TENSORcnt-tracked DMA), scores never touch HBM.
// ---------------------------------------------------------------------------

typedef __attribute__((ext_vector_type(16))) __bf16 v16bf;
typedef __attribute__((ext_vector_type(8)))  float  v8f;
typedef __attribute__((ext_vector_type(4)))  unsigned int v4u;
typedef __attribute__((ext_vector_type(8)))  int v8i;
typedef __attribute__((ext_vector_type(4)))  int v4i;

#define B_   16
#define C_   1024
#define MID_ 256
#define HW_  2304   // 48*48

__device__ __forceinline__ unsigned short f2bf(float f) {
  unsigned int u = __float_as_uint(f);
  u += 0x7FFFu + ((u >> 16) & 1u);   // round-to-nearest-even
  return (unsigned short)(u >> 16);
}

union V16 {
  v16bf v;
  unsigned int u[8];
};

// A-matrix (16x32 bf16) load, row-major source with row stride `rowStride`
// (elements). ISA 16-bit A layout: lanes 0-15 K={0..7,16..23}, lanes 16-31
// K={8..15,24..31}, 2 K per VGPR.
__device__ __forceinline__ V16 loadA16(const unsigned short* base, int rowStride, int lane) {
  V16 r;
  int M = lane & 15, kh = lane >> 4;
  const unsigned short* row = base + M * rowStride;
#pragma unroll
  for (int p = 0; p < 8; ++p) {
    int kb = ((p < 4) ? 0 : 16) + kh * 8 + (p & 3) * 2;
    r.u[p] = *(const unsigned int*)(row + kb);
  }
  return r;
}

// B-matrix (32x16 bf16) load, element (K,N) at base + N*nStride + K.
__device__ __forceinline__ V16 loadB16(const unsigned short* base, int nStride, int lane) {
  V16 r;
  int N = lane & 15, kh = lane >> 4;
  const unsigned short* col = base + N * nStride + kh * 8;
#pragma unroll
  for (int p = 0; p < 8; ++p) r.u[p] = *(const unsigned int*)(col + ((p < 4) ? 0 : 16) + (p & 3) * 2);
  return r;
}

__device__ __forceinline__ v8f wmma_bf16(const V16& a, const V16& b, v8f c) {
  return __builtin_amdgcn_wmma_f32_16x16x32_bf16(
      false, a.v, false, b.v, (short)0, c, false, false);
}

// ---------------------------------------------------------------------------
// Tensor Data Mover: 2D tile load Global -> LDS (D# per ISA 08_async_tensor).
// data_size_code: 0=1B 1=2B 2=4B 3=8B. Dims/strides in data_size units.
// Toolchain uses the 6-arg builtin form (clang-23 / therock headers):
//   (uint32x4 g0, int32x8 g1, int32x4 g2, int32x4 g3, int32x8, i32 cpol)
// ---------------------------------------------------------------------------
#if defined(__gfx1250__) && __has_builtin(__builtin_amdgcn_tensor_load_to_lds)
#define HAVE_TDM 1
__device__ __forceinline__ void tdm_load_2d(unsigned lds_off, const void* gptr,
                                            unsigned tile_d0, unsigned tile_d1,
                                            unsigned tensor_d0, unsigned tensor_d1,
                                            unsigned long long d0_stride,
                                            unsigned data_size_code) {
  unsigned long long ga = (unsigned long long)(__SIZE_TYPE__)gptr;
  v4u g0;
  g0[0] = 1u;                                                // count=1, user D#
  g0[1] = lds_off;                                           // lds_addr (bytes)
  g0[2] = (unsigned)(ga & 0xFFFFFFFFu);                      // global_addr lo
  g0[3] = (unsigned)((ga >> 32) & 0x01FFFFFFu) | (2u << 30); // addr hi | type=2
  v8i g1;
  g1[0] = (int)(data_size_code << 16);                       // wg_mask=0
  g1[1] = (int)((tensor_d0 & 0xFFFFu) << 16);                // tensor_dim0 lo
  g1[2] = (int)((tensor_d0 >> 16) | ((tensor_d1 & 0xFFFFu) << 16));
  g1[3] = (int)((tensor_d1 >> 16) | (tile_d0 << 16));        // tile_dim0
  g1[4] = (int)(tile_d1 & 0xFFFFu);                          // tile_dim1, dim2=0
  g1[5] = (int)(d0_stride & 0xFFFFFFFFu);                    // dim0_stride lo
  g1[6] = (int)((d0_stride >> 32) & 0xFFFFu);                // dim0_stride hi
  g1[7] = 0;
  v4i z4 = {0, 0, 0, 0};
  v8i z8 = {0, 0, 0, 0, 0, 0, 0, 0};
  __builtin_amdgcn_tensor_load_to_lds(g0, g1, z4, z4, z8, 0);
}
#else
#define HAVE_TDM 0
#endif

// ---------------------------------------------------------------------------
// Weight conversion: optional per-row BN scale fold, f32 -> bf16
// ---------------------------------------------------------------------------
__global__ void wfold_kernel(const float* __restrict__ W,
                             const float* __restrict__ gamma,
                             const float* __restrict__ var,
                             unsigned short* __restrict__ dst,
                             int rows, int cols) {
  int idx = blockIdx.x * blockDim.x + threadIdx.x;
  if (idx >= rows * cols) return;
  float s = 1.0f;
  if (gamma) {
    int r = idx / cols;
    s = gamma[r] * rsqrtf(var[r] + 1e-5f);
  }
  dst[idx] = f2bf(W[idx] * s);
}

__global__ void bfold_kernel(const float* __restrict__ b,
                             const float* __restrict__ gamma,
                             const float* __restrict__ beta,
                             const float* __restrict__ mean,
                             const float* __restrict__ var,
                             float* __restrict__ dst, int n) {
  int i = blockIdx.x * blockDim.x + threadIdx.x;
  if (i >= n) return;
  float inv = gamma[i] * rsqrtf(var[i] + 1e-5f);
  dst[i] = b[i] * inv + beta[i] - mean[i] * inv;
}

// ---------------------------------------------------------------------------
// Projection kernel: f/g/h = W' @ x  (per batch, K = C = 1024)
// ---------------------------------------------------------------------------
__global__ void __launch_bounds__(128) proj_kernel(
    const float* __restrict__ x,
    const unsigned short* __restrict__ Wf16,
    const unsigned short* __restrict__ Wg16,
    const unsigned short* __restrict__ Wh16,
    const float* __restrict__ biasF,
    const float* __restrict__ biasG,
    const float* __restrict__ bh,
    unsigned short* __restrict__ fT,   // [B][HW][mid]
    unsigned short* __restrict__ gT,   // [B][HW][mid]
    unsigned short* __restrict__ hM)   // [B][mid][HW]
{
  __shared__ unsigned short xs[16 * 32];  // [nn][kk], K contiguous per N

  int bn = blockIdx.x;
  int b  = bn / (HW_ / 16);
  int n0 = (bn % (HW_ / 16)) * 16;
  int wid  = threadIdx.x >> 5;
  int lane = threadIdx.x & 31;
  int g    = blockIdx.y * 4 + wid;   // 0..47
  int proj = g >> 4;                 // 0:f 1:g 2:h
  int m0   = (g & 15) * 16;

  const unsigned short* Wsel = (proj == 0) ? Wf16 : (proj == 1) ? Wg16 : Wh16;
  const float* bsel          = (proj == 0) ? biasF : (proj == 1) ? biasG : bh;

  const float* xb = x + (size_t)b * C_ * HW_;

  v8f acc = {};
  int nn = threadIdx.x & 15;   // staging: contiguous lanes -> contiguous n
  int kg = threadIdx.x >> 4;   // 0..7

  for (int kc = 0; kc < C_ / 32; ++kc) {
#pragma unroll
    for (int i = 0; i < 4; ++i) {
      int kk = kg + 8 * i;
      float v = xb[(size_t)(kc * 32 + kk) * HW_ + n0 + nn];
      xs[nn * 32 + kk] = f2bf(v);
    }
    if (kc + 1 < C_ / 32) {
      __builtin_prefetch(&xb[(size_t)((kc + 1) * 32 + kg) * HW_ + n0 + nn], 0, 1);
    }
    __syncthreads();
    V16 a = loadA16(Wsel + m0 * C_ + kc * 32, C_, lane);
    V16 bm = loadB16(xs, 32, lane);
    acc = wmma_bf16(a, bm, acc);
    __syncthreads();
  }

  int Nn = lane & 15, half = lane >> 4;
#pragma unroll
  for (int v = 0; v < 8; ++v) {
    int m = m0 + v + half * 8;
    int n = n0 + Nn;
    float y = acc[v] + bsel[m];
    if (proj == 0) {
      y = fmaxf(y, 0.0f);
      fT[((size_t)b * HW_ + n) * MID_ + m] = f2bf(y);
    } else if (proj == 1) {
      y = fmaxf(y, 0.0f);
      gT[((size_t)b * HW_ + n) * MID_ + m] = f2bf(y);
    } else {
      hM[((size_t)b * MID_ + m) * HW_ + n] = f2bf(y);
    }
  }
}

// ---------------------------------------------------------------------------
// Flash attention. Block = 4 waves = 4 consecutive i-tiles (always same batch:
// 144 i-tiles per batch, 4 | 144). Per 32-column j-step:
//   - wave 0 DMAs K tile (32j x 256c) and V tile (256c x 32j) bf16 into LDS
//     via TENSOR_LOAD_TO_LDS, waits TENSORcnt, barrier releases the block
//   - 16 WMMAs for S, online softmax (shfl_xor row reductions in 16-lane
//     halves), P f32->bf16 through per-wave LDS (C-layout -> A-layout)
//   - 16 WMMAs for O += P V
// ---------------------------------------------------------------------------
__global__ void __launch_bounds__(128) attn_kernel(
    const unsigned short* __restrict__ fT,
    const unsigned short* __restrict__ gT,
    const unsigned short* __restrict__ hM,
    unsigned short* __restrict__ zT)   // [B][HW][mid]
{
  __shared__ unsigned short ldsK[32 * MID_];   // 16 KB: [j 0..31][c 0..255]
  __shared__ unsigned short ldsV[MID_ * 32];   // 16 KB: [c 0..255][j 0..31]
  __shared__ unsigned short ps[4][16 * 32];    //  4 KB: per-wave P staging

  int wid = threadIdx.x >> 5, lane = threadIdx.x & 31;
  int itile = blockIdx.x * 4 + wid;
  int b  = itile / (HW_ / 16);
  int i0 = (itile % (HW_ / 16)) * 16;

  const unsigned short* fTb = fT + (size_t)b * HW_ * MID_;
  const unsigned short* gTb = gT + (size_t)b * HW_ * MID_;
  const unsigned short* hb  = hM + (size_t)b * MID_ * HW_;
  unsigned short* zTb       = zT + (size_t)b * HW_ * MID_;

  // Preload Q: 8 A-tiles covering d=256
  V16 q[8];
#pragma unroll
  for (int c8 = 0; c8 < 8; ++c8)
    q[c8] = loadA16(fTb + (size_t)i0 * MID_ + c8 * 32, MID_, lane);

  v8f O[16];
#pragma unroll
  for (int t = 0; t < 16; ++t) O[t] = {};
  float mrow[8], lrow[8];
#pragma unroll
  for (int v = 0; v < 8; ++v) { mrow[v] = -3.0e38f; lrow[v] = 0.0f; }

  const float scale = 0.0625f;  // mid^-0.5 = 1/16
  int Nn = lane & 15, half = lane >> 4;

#if HAVE_TDM
  unsigned ldsK_off = (unsigned)(__SIZE_TYPE__)(void*)ldsK;
  unsigned ldsV_off = (unsigned)(__SIZE_TYPE__)(void*)ldsV;
#endif

  for (int j0 = 0; j0 < HW_; j0 += 32) {
#if HAVE_TDM
    if (wid == 0) {
      // K: rows j0..j0+31 of gT (row = 256 bf16, rows contiguous)
      tdm_load_2d(ldsK_off, gTb + (size_t)j0 * MID_, MID_, 32, MID_, HW_,
                  MID_, 1u);
      // V: h[c][j0..j0+32), 256 rows of 32 bf16, row stride HW
      tdm_load_2d(ldsV_off, hb + j0, 32, MID_, HW_, MID_, HW_, 1u);
      __builtin_amdgcn_s_wait_tensorcnt(0);
    }
#else
    {
      // Cooperative fallback copy Global -> LDS (128 threads)
      const uint4* srcK = (const uint4*)(gTb + (size_t)j0 * MID_);
      uint4* dstK = (uint4*)ldsK;
#pragma unroll
      for (int i = 0; i < 8; ++i) dstK[threadIdx.x + 128 * i] = srcK[threadIdx.x + 128 * i];
#pragma unroll
      for (int i = 0; i < 2; ++i) {
        int c = threadIdx.x + 128 * i;
        const uint4* srcV = (const uint4*)(hb + (size_t)c * HW_ + j0);
        uint4* dstV = (uint4*)(ldsV + c * 32);
        dstV[0] = srcV[0]; dstV[1] = srcV[1]; dstV[2] = srcV[2]; dstV[3] = srcV[3];
      }
    }
#endif
    __syncthreads();

    v8f S0 = {}, S1 = {};
#pragma unroll
    for (int c8 = 0; c8 < 8; ++c8) {
      // S0: B[K=c][N=j(0..15)] = ldsK[N*256 + c8*32 + K]
      V16 bk = loadB16(ldsK + c8 * 32, MID_, lane);
      S0 = wmma_bf16(q[c8], bk, S0);
    }
#pragma unroll
    for (int c8 = 0; c8 < 8; ++c8) {
      V16 bk = loadB16(ldsK + 16 * MID_ + c8 * 32, MID_, lane);
      S1 = wmma_bf16(q[c8], bk, S1);
    }

    float alpha[8];
#pragma unroll
    for (int v = 0; v < 8; ++v) {
      float s0 = S0[v] * scale, s1 = S1[v] * scale;
      float r = fmaxf(s0, s1);                 // row max across 16-lane half
      r = fmaxf(r, __shfl_xor(r, 1, 32));
      r = fmaxf(r, __shfl_xor(r, 2, 32));
      r = fmaxf(r, __shfl_xor(r, 4, 32));
      r = fmaxf(r, __shfl_xor(r, 8, 32));
      float mnew = fmaxf(mrow[v], r);
      float a = __expf(mrow[v] - mnew);
      mrow[v] = mnew;
      float p0 = __expf(s0 - mnew), p1 = __expf(s1 - mnew);
      float rs = p0 + p1;
      rs += __shfl_xor(rs, 1, 32);
      rs += __shfl_xor(rs, 2, 32);
      rs += __shfl_xor(rs, 4, 32);
      rs += __shfl_xor(rs, 8, 32);
      lrow[v] = lrow[v] * a + rs;
      alpha[v] = a;
      int row = v + half * 8;                  // C-layout row for this VGPR
      ps[wid][row * 32 + Nn]      = f2bf(p0);
      ps[wid][row * 32 + 16 + Nn] = f2bf(p1);
    }

#pragma unroll
    for (int t = 0; t < 16; ++t)
#pragma unroll
      for (int v = 0; v < 8; ++v) O[t][v] *= alpha[v];

    __syncthreads();
    V16 Pa = loadA16(ps[wid], 32, lane);       // A-layout P (16x32 bf16)
#pragma unroll
    for (int ct = 0; ct < 16; ++ct) {
      // B[K=j][N=c] = ldsV[(ct*16+N)*32 + K]
      V16 bv = loadB16(ldsV + ct * 16 * 32, 32, lane);
      O[ct] = wmma_bf16(Pa, bv, O[ct]);
    }
    __syncthreads();   // protect ldsK/ldsV/ps before next step's refill
  }

#pragma unroll
  for (int v = 0; v < 8; ++v) {
    float inv = 1.0f / lrow[v];
    int row = i0 + v + half * 8;
#pragma unroll
    for (int t = 0; t < 16; ++t) {
      float o = O[t][v] * inv;
      zTb[(size_t)row * MID_ + t * 16 + Nn] = f2bf(o);
    }
  }
}

// ---------------------------------------------------------------------------
// Output 1x1 conv + bias + residual: out = Wv @ z + bv + x (K = mid = 256)
// ---------------------------------------------------------------------------
__global__ void __launch_bounds__(128) outconv_kernel(
    const unsigned short* __restrict__ Wv16,   // [C][mid]
    const unsigned short* __restrict__ zT,     // [B][HW][mid]
    const float* __restrict__ bv,
    const float* __restrict__ x,
    float* __restrict__ out)
{
  int wid = threadIdx.x >> 5, lane = threadIdx.x & 31;
  int bm = blockIdx.x;              // B*64
  int b  = bm >> 6;
  int o0 = (bm & 63) * 16;
  int n0 = (blockIdx.y * 4 + wid) * 16;
  const unsigned short* zTb = zT + (size_t)b * HW_ * MID_;

  v8f acc = {};
#pragma unroll
  for (int c8 = 0; c8 < 8; ++c8) {
    V16 a  = loadA16(Wv16 + o0 * MID_ + c8 * 32, MID_, lane);
    V16 bb = loadB16(zTb + (size_t)n0 * MID_ + c8 * 32, MID_, lane);
    acc = wmma_bf16(a, bb, acc);
  }

  int Nn = lane & 15, half = lane >> 4;
#pragma unroll
  for (int v = 0; v < 8; ++v) {
    int o = o0 + v + half * 8;
    size_t idx = ((size_t)b * C_ + o) * HW_ + n0 + Nn;   // coalesced
    out[idx] = acc[v] + bv[o] + x[idx];
  }
}

// ---------------------------------------------------------------------------
extern "C" void kernel_launch(void* const* d_in, const int* in_sizes, int n_in,
                              void* d_out, int out_size, void* d_ws, size_t ws_size,
                              hipStream_t stream) {
  const float* x   = (const float*)d_in[0];
  const float* Wf  = (const float*)d_in[1];
  const float* bf  = (const float*)d_in[2];
  const float* gf  = (const float*)d_in[3];
  const float* btf = (const float*)d_in[4];
  const float* mf  = (const float*)d_in[5];
  const float* vf  = (const float*)d_in[6];
  const float* Wg  = (const float*)d_in[7];
  const float* bg  = (const float*)d_in[8];
  const float* gg  = (const float*)d_in[9];
  const float* btg = (const float*)d_in[10];
  const float* mg  = (const float*)d_in[11];
  const float* vg  = (const float*)d_in[12];
  const float* Wh  = (const float*)d_in[13];
  const float* bh  = (const float*)d_in[14];
  const float* Wv  = (const float*)d_in[15];
  const float* bvp = (const float*)d_in[16];
  float* out = (float*)d_out;

  char* ws = (char*)d_ws;
  const size_t WSZ = (size_t)256 * 1024 * 2;           // 512 KB per weight
  unsigned short* Wf16 = (unsigned short*)(ws);
  unsigned short* Wg16 = (unsigned short*)(ws + WSZ);
  unsigned short* Wh16 = (unsigned short*)(ws + 2 * WSZ);
  unsigned short* Wv16 = (unsigned short*)(ws + 3 * WSZ);
  float* biasF = (float*)(ws + 4 * WSZ);
  float* biasG = (float*)(ws + 4 * WSZ + 1024);
  unsigned short* fT = (unsigned short*)(ws + 4 * WSZ + 2048);
  unsigned short* gT = fT + (size_t)B_ * HW_ * MID_;
  unsigned short* hM = gT + (size_t)B_ * HW_ * MID_;
  unsigned short* zT = hM + (size_t)B_ * HW_ * MID_;

  int wfoldN = 256 * 1024;
  wfold_kernel<<<(wfoldN + 255) / 256, 256, 0, stream>>>(Wf, gf, vf, Wf16, 256, 1024);
  wfold_kernel<<<(wfoldN + 255) / 256, 256, 0, stream>>>(Wg, gg, vg, Wg16, 256, 1024);
  wfold_kernel<<<(wfoldN + 255) / 256, 256, 0, stream>>>(Wh, nullptr, nullptr, Wh16, 256, 1024);
  wfold_kernel<<<(wfoldN + 255) / 256, 256, 0, stream>>>(Wv, nullptr, nullptr, Wv16, 1024, 256);
  bfold_kernel<<<1, 256, 0, stream>>>(bf, gf, btf, mf, vf, biasF, 256);
  bfold_kernel<<<1, 256, 0, stream>>>(bg, gg, btg, mg, vg, biasG, 256);

  dim3 pg(B_ * (HW_ / 16), 12);
  proj_kernel<<<pg, 128, 0, stream>>>(x, Wf16, Wg16, Wh16, biasF, biasG, bh, fT, gT, hM);

  attn_kernel<<<(B_ * (HW_ / 16)) / 4, 128, 0, stream>>>(fT, gT, hM, zT);

  dim3 og(B_ * 64, 36);
  outconv_kernel<<<og, 128, 0, stream>>>(Wv16, zT, bvp, x, out);
}
